// GraphCNN_3882650436681
// MI455X (gfx1250) — matrix operand order repaired and checked by reading, not compiled, not run on hardware
//
#include <hip/hip_runtime.h>
#include <hip/hip_bf16.h>
#include <stdint.h>

// ---------------------------------------------------------------------------
// GCN on MI455X (gfx1250, wave32).
// GEMMs: v_wmma_f32_16x16x32_bf16, dual accumulator chains so independent
// WMMAs can issue back-to-back (no D->C hazard NOPs between them). Operands
// pre-staged in bf16 -> inner loop is b128 A loads + strided 16-bit B loads + WMMA.
// Aggregation: native global_atomic_add_f32 scatter; h/agg (51.2 MB each)
// are L2-resident (192 MB L2), so the scatter is L2-BW bound.
// ---------------------------------------------------------------------------

typedef __attribute__((ext_vector_type(16))) __bf16 v16bf;
typedef __attribute__((ext_vector_type(8)))  float  v8f;

union Frag16 {
  v16bf v;
  unsigned short u[16];
  uint4 q[2];          // two 128-bit halves for vector loads
};

__device__ __forceinline__ unsigned short f2bf(float f) {
  unsigned u = __float_as_uint(f);
  u += 0x7FFFu + ((u >> 16) & 1u);          // round-to-nearest-even
  return (unsigned short)(u >> 16);
}

__device__ __forceinline__ void atomAddF(float* p, float v) {
  unsafeAtomicAdd(p, v);                    // -> global_atomic_add_f32
}

// ---------------- small utility kernels ----------------

// convert K x C f32 weights into K x Cp bf16 (zero-padded columns)
__global__ void k_w_to_bf16_pad(const float* __restrict__ in,
                                unsigned short* __restrict__ out,
                                int K, int C, int Cp) {
  int i = blockIdx.x * blockDim.x + threadIdx.x;
  if (i < K * Cp) {
    int k = i / Cp, c = i % Cp;
    out[i] = (c < C) ? f2bf(in[(unsigned)k * C + c]) : (unsigned short)0;
  }
}

// dense f32 -> bf16, 4 elements/thread (b128 load, b64 store)
__global__ void k_f32_to_bf16_v4(const float* __restrict__ in,
                                 unsigned short* __restrict__ out, int n4) {
  int i = blockIdx.x * blockDim.x + threadIdx.x;
  if (i < n4) {
    float4 f = ((const float4*)in)[i];
    ushort4 o;
    o.x = f2bf(f.x); o.y = f2bf(f.y); o.z = f2bf(f.z); o.w = f2bf(f.w);
    ((ushort4*)out)[i] = o;
  }
}

__global__ void k_deg_init(float* __restrict__ deg, int n) {
  int i = blockIdx.x * blockDim.x + threadIdx.x;
  if (i < n) deg[i] = 1.0f;                 // self-loop contributes 1
}

__global__ void k_deg_count(const long long* __restrict__ dst,
                            float* __restrict__ deg, int E) {
  int e = blockIdx.x * blockDim.x + threadIdx.x;
  if (e < E) atomAddF(&deg[(int)dst[e]], 1.0f);
}

__global__ void k_deg_rsqrt(float* __restrict__ deg, int n) {
  int i = blockIdx.x * blockDim.x + threadIdx.x;
  if (i < n) { float d = deg[i]; deg[i] = d > 0.f ? rsqrtf(d) : 0.f; }
}

// ---------------- WMMA GEMM: D[MxNcol] = A[Mx128](bf16) * W[128xNcolP](bf16) ----
// One wave computes one 16x16 output tile; K=128 -> 4 x v_wmma_f32_16x16x32_bf16
// on two independent accumulator chains (k-halves interleaved), summed at end.
// Fragment layouts per ISA 7.12.2. REQUIRES M % 16 == 0 (launcher guarantees;
// no per-row store guards -> straight-line epilogue).
// NcolP: padded (multiple-of-16) stride of Wbf; Ncol: true output width.
__global__ void k_gemm_wmma_bf16(const unsigned short* __restrict__ A,
                                 const unsigned short* __restrict__ Wbf,
                                 float* __restrict__ D,
                                 int Ncol, int NcolP) {
  const int lane = threadIdx.x;             // 0..31
  const int half = lane >> 4;
  const int l15  = lane & 15;
  const int rowBase = blockIdx.x * 16;
  const int colBase = blockIdx.y * 16;

  const unsigned short* Arow = A + (unsigned)(rowBase + l15) * 128u;

  v8f c0 = {}, c1 = {};
  #pragma unroll
  for (int k0 = 0; k0 < 128; k0 += 64) {
    // chain 0: K block [k0, k0+32)
    Frag16 a0, b0;
    {
      const uint4* ap = (const uint4*)(Arow + k0 + half * 8);
      a0.q[0] = ap[0];                  // K = k0 + base .. +7
      a0.q[1] = ap[2];                  // K = k0 + base+16 .. +23
      const unsigned kk = (unsigned)(k0 + half * 8);
      const unsigned short* wp = Wbf + kk * (unsigned)NcolP + colBase + l15;
      #pragma unroll
      for (int s = 0; s < 16; ++s)
        b0.u[s] = wp[(unsigned)((s & 7) + (s >> 3) * 16) * (unsigned)NcolP];
    }
    // chain 1: K block [k0+32, k0+64)
    Frag16 a1, b1;
    {
      const uint4* ap = (const uint4*)(Arow + k0 + 32 + half * 8);
      a1.q[0] = ap[0];
      a1.q[1] = ap[2];
      const unsigned kk = (unsigned)(k0 + 32 + half * 8);
      const unsigned short* wp = Wbf + kk * (unsigned)NcolP + colBase + l15;
      #pragma unroll
      for (int s = 0; s < 16; ++s)
        b1.u[s] = wp[(unsigned)((s & 7) + (s >> 3) * 16) * (unsigned)NcolP];
    }
    c0 = __builtin_amdgcn_wmma_f32_16x16x32_bf16(false, a0.v, false, b0.v,
                                                 (short)0, c0, false, false);
    c1 = __builtin_amdgcn_wmma_f32_16x16x32_bf16(false, a1.v, false, b1.v,
                                                 (short)0, c1, false, false);
  }

  // C/D layout: lanes 0-15: (M=rowBase+v, N=colBase+lane); lanes 16-31: M+=8.
  const int col = colBase + l15;
  if (col < Ncol) {
    float* dp = D + (unsigned)(rowBase + half * 8) * (unsigned)Ncol + col;
    #pragma unroll
    for (int v = 0; v < 8; ++v)
      dp[(unsigned)v * (unsigned)Ncol] = c0[v] + c1[v];
  }
}

// ---------------- aggregation ----------------

// agg[i,:] = h[i,:] * dinv[i]^2  (self-loop term; doubles as the zero-init).
// float4 per thread; C % 4 == 0 so a float4 never straddles rows.
__global__ void k_selfloop_init_v4(const float* __restrict__ h,
                                   const float* __restrict__ dinv,
                                   float* __restrict__ agg, int total4, int C) {
  int i = blockIdx.x * blockDim.x + threadIdx.x;
  if (i < total4) {
    int node = (i * 4) / C;
    float s = dinv[node]; s *= s;
    float4 v = ((const float4*)h)[i];
    v.x *= s; v.y *= s; v.z *= s; v.w *= s;
    ((float4*)agg)[i] = v;
  }
}

// agg[dst,:] += h[src,:] * dinv[src]*dinv[dst]   — one edge per block,
// threadIdx.x = channel (coalesced reads, native f32 atomics into L2).
__global__ void k_edge_agg(const long long* __restrict__ src,
                           const long long* __restrict__ dst,
                           const float* __restrict__ dinv,
                           const float* __restrict__ h,
                           float* __restrict__ agg, int C) {
  const int e = blockIdx.x;
  const int c = threadIdx.x;
  const int s = (int)src[e];
  const int d = (int)dst[e];
  const float nrm = dinv[s] * dinv[d];
  if (c < C)
    atomAddF(&agg[(unsigned)d * (unsigned)C + c], h[(unsigned)s * (unsigned)C + c] * nrm);
}

// hid_bf16 = bf16(relu(agg + b)) — feeds GEMM2's A operand directly.
// float4 per thread (C % 4 == 0).
__global__ void k_bias_relu_bf16_v4(const float* __restrict__ agg,
                                    const float* __restrict__ b,
                                    unsigned short* __restrict__ hidbf,
                                    int total4, int C) {
  int i = blockIdx.x * blockDim.x + threadIdx.x;
  if (i < total4) {
    int c0 = (i * 4) % C;
    float4 v  = ((const float4*)agg)[i];
    float4 bv = *(const float4*)(b + c0);
    float x0 = v.x + bv.x, x1 = v.y + bv.y, x2 = v.z + bv.z, x3 = v.w + bv.w;
    ushort4 o;
    o.x = f2bf(x0 > 0.f ? x0 : 0.f);
    o.y = f2bf(x1 > 0.f ? x1 : 0.f);
    o.z = f2bf(x2 > 0.f ? x2 : 0.f);
    o.w = f2bf(x3 > 0.f ? x3 : 0.f);
    ((ushort4*)hidbf)[i] = o;
  }
}

// fused: out[row,:] = log_softmax(out[row,:] + b2).  One wave per 40-wide row.
__global__ void k_bias_logsoftmax40(float* __restrict__ out,
                                    const float* __restrict__ b2, int n) {
  const int wid = threadIdx.x >> 5, lane = threadIdx.x & 31;
  const int row = blockIdx.x * 8 + wid;
  if (row >= n) return;
  float* r = out + (unsigned)row * 40u;
  float v0 = r[lane] + b2[lane];
  float v1 = (lane < 8) ? (r[32 + lane] + b2[32 + lane]) : -__builtin_inff();
  float m = fmaxf(v0, v1);
  #pragma unroll
  for (int off = 16; off; off >>= 1) m = fmaxf(m, __shfl_xor(m, off, 32));
  float s = __expf(v0 - m) + ((lane < 8) ? __expf(v1 - m) : 0.f);
  #pragma unroll
  for (int off = 16; off; off >>= 1) s += __shfl_xor(s, off, 32);
  const float ls = m + __logf(s);
  r[lane] = v0 - ls;
  if (lane < 8) r[32 + lane] = v1 - ls;
}

// ---------------- host launcher ----------------

extern "C" void kernel_launch(void* const* d_in, const int* in_sizes, int n_in,
                              void* d_out, int out_size, void* d_ws, size_t ws_size,
                              hipStream_t stream) {
  const float*     x   = (const float*)d_in[0];
  const long long* ei  = (const long long*)d_in[1];   // int64 edge_index [2,E]
  const float*     W1  = (const float*)d_in[2];
  const float*     b1  = (const float*)d_in[3];
  const float*     W2  = (const float*)d_in[4];
  const float*     b2  = (const float*)d_in[5];

  const int IN_C = 128, HID_C = 128, OUT_C = 40;
  const int OUT_CP = 48;                    // padded column count for WMMA tiles
  const int N = in_sizes[0] / IN_C;         // 100000 (divisible by 16)
  const int E = in_sizes[1] / 2;
  const long long* src = ei;
  const long long* dst = ei + E;

  // workspace: dinv | W1bf | W2bf(pad) | abf (xbf then hidbf) | h1(=h2) | agg1
  char* ws = (char*)d_ws;
  size_t off = 0;
  auto carve = [&](size_t bytes) -> void* {
    void* p = ws + off;
    off = (off + bytes + 511) & ~(size_t)511;
    return p;
  };
  float*          dinv = (float*)         carve((size_t)N * 4);
  unsigned short* W1bf = (unsigned short*)carve((size_t)IN_C * HID_C * 2);
  unsigned short* W2bf = (unsigned short*)carve((size_t)HID_C * OUT_CP * 2);
  unsigned short* abf  = (unsigned short*)carve((size_t)N * HID_C * 2); // xbf / hidbf
  float*          h1   = (float*)         carve((size_t)N * HID_C * 4);
  float*          agg1 = (float*)         carve((size_t)N * HID_C * 4);
  float*          h2   = h1;              // h1 is dead once GEMM2 runs
  float*          out  = (float*)d_out;   // doubles as agg2

  const int Mtiles = N / 16;              // exact: N % 16 == 0
  const int n4_hid = (N * HID_C) / 4;
  const int n4_out = (N * OUT_C) / 4;

  // weights -> bf16 (W2 zero-padded 40 -> 48 columns)
  k_w_to_bf16_pad<<<(IN_C * HID_C + 255) / 256, 256, 0, stream>>>(W1, W1bf, IN_C, HID_C, HID_C);
  k_w_to_bf16_pad<<<(HID_C * OUT_CP + 255) / 256, 256, 0, stream>>>(W2, W2bf, HID_C, OUT_C, OUT_CP);

  // degrees (incl. self-loop) -> dinv = rsqrt(deg), in place
  k_deg_init <<<(N + 255) / 256, 256, 0, stream>>>(dinv, N);
  k_deg_count<<<(E + 255) / 256, 256, 0, stream>>>(dst, dinv, E);
  k_deg_rsqrt<<<(N + 255) / 256, 256, 0, stream>>>(dinv, N);

  // layer 1: xbf = bf16(x); h1 = xbf @ W1 ; agg1 = A_norm @ h1
  k_f32_to_bf16_v4<<<(n4_hid + 255) / 256, 256, 0, stream>>>(x, abf, n4_hid);
  k_gemm_wmma_bf16<<<dim3(Mtiles, HID_C / 16), 32, 0, stream>>>(abf, W1bf, h1, HID_C, HID_C);
  k_selfloop_init_v4<<<(n4_hid + 255) / 256, 256, 0, stream>>>(h1, dinv, agg1, n4_hid, HID_C);
  k_edge_agg<<<E, 128, 0, stream>>>(src, dst, dinv, h1, agg1, HID_C);

  // hidbf = bf16(relu(agg1 + b1))  (GEMM2 consumes bf16 directly)
  k_bias_relu_bf16_v4<<<(n4_hid + 255) / 256, 256, 0, stream>>>(agg1, b1, abf, n4_hid, HID_C);

  // layer 2: h2 = hidbf @ W2 ; out = A_norm @ h2 ; out = log_softmax(out + b2)
  k_gemm_wmma_bf16<<<dim3(Mtiles, OUT_CP / 16), 32, 0, stream>>>(abf, W2bf, h2, OUT_C, OUT_CP);
  k_selfloop_init_v4<<<(n4_out + 255) / 256, 256, 0, stream>>>(h2, dinv, out, n4_out, OUT_C);
  k_edge_agg<<<E, 64, 0, stream>>>(src, dst, dinv, h2, out, OUT_C);
  k_bias_logsoftmax40<<<(N + 7) / 8, 256, 0, stream>>>(out, b2, N);
}
